// GNN_20813411517053
// MI455X (gfx1250) — compile-verified, matrix operand order checked by reading, not run on hardware
//
#include <hip/hip_runtime.h>
#include <hip/hip_bf16.h>
#include <math.h>

// ---------------------------------------------------------------------------
// MI455X (gfx1250) implementation. All channel-GEMMs (1x1 convs, 3x3 convs as
// 9-tap implicit GEMM, 256->10 context projection) run on
// v_wmma_f32_16x16x32_bf16 (bf16 A/B, f32 accumulate). Weights are staged in
// LDS in A-fragment layout once per block; compile-time channel/tap counts
// fold all predicates; BN(training) stats accumulate via LDS float atomics
// (ds_add_f32) and two global atomics (sum, sumsq) per channel per block.
// Grid requirement: B*HW divisible by 64 (8*9216 = 73728 ✓).
// ---------------------------------------------------------------------------

typedef __attribute__((ext_vector_type(16))) __bf16 bf16x16;
typedef __attribute__((ext_vector_type(8)))  float  f32x8;

__device__ __forceinline__ __bf16 f2bf(float f) {
  unsigned u = __builtin_bit_cast(unsigned, f);
  unsigned short h = (unsigned short)((u + 0x7FFFu + ((u >> 16) & 1u)) >> 16);
  return __builtin_bit_cast(__bf16, h);
}

// 16-bit A/B fragment K index for element e (0..15) in lane-half hi (ISA 7.12.2)
__device__ __forceinline__ constexpr int kmapK(int e, int hi) {
  return ((e < 8) ? 0 : 16) + (hi ? 8 : 0) + (e & 7);
}

struct Src {                 // strided f32 source with optional multiplicative gate
  const float* x; long sC, sB;      // addr = c*sC + b*sB + hw
  const float* s; long ssC, ssB;    // gate addr = c*ssC + b*ssB + hw (null = 1.0)
};

struct GemmP {
  const float* W;            // [Cout, Cin, taps]
  const float* bias;         // [Cout] or null
  int Cout, C0, C1, taps;    // runtime copies (dispatch key); templates hold truth
  Src s0, s1;
  float* Y; long ysC, ysB;   // output strides (addr = c*ysC + b*ysB + hw)
  float* stats;              // [128]: sum at [c], sumsq at [64+c]
  int acc, act;              // runtime copies
  int B, HW, Wimg;
};

template <int MT, int C0, int C1, int TAPS, bool SC0, bool SC1, bool STATS, int ACT>
__global__ __launch_bounds__(128) void k_gemm(GemmP p) {
  constexpr int CIN = C0 + C1;
  constexpr int KS  = (CIN + 31) / 32;
  constexpr int NFRAG = TAPS * KS * MT;
  __shared__ bf16x16 wlds[NFRAG * 32];   // A fragments: [frag][lane] -> 16 bf16
  __shared__ float bstat[128];           // block-level BN partials

  const int tid  = threadIdx.x;
  const int lane = tid & 31;
  const int wid  = tid >> 5;

  // ---- stage weights into LDS in exact A-fragment layout (whole block) ----
  {
    __bf16* wl = reinterpret_cast<__bf16*>(wlds);
    const int total = NFRAG * 512;
    for (int i = tid; i < total; i += 128) {
      int e    = i & 15;
      int ln2  = (i >> 4) & 15;
      int hi2  = (i >> 8) & 1;
      int frag = i >> 9;
      int mt   = frag % MT;
      int rest = frag / MT;
      int kk   = rest % KS;
      int tap  = rest / KS;
      int m = mt * 16 + ln2;
      int k = kk * 32 + kmapK(e, hi2);
      float w = (m < p.Cout && k < CIN)
                    ? p.W[((long)m * CIN + k) * TAPS + tap] : 0.f;
      wl[i] = f2bf(w);
    }
  }
  if (STATS) { if (tid < 128) bstat[tid] = 0.f; }
  __syncthreads();

  const int tile = blockIdx.x * 4 + wid;        // grid covers P/16 tiles exactly
  const Src s0 = p.s0, s1 = p.s1;
  const int hi  = lane >> 4;
  const int ln  = lane & 15;
  const int pix = tile * 16 + ln;               // pixel (N) for B-frag / store
  const int b   = pix / p.HW;
  const int hw  = pix % p.HW;
  const int Hi  = p.HW / p.Wimg;
  const int yy  = hw / p.Wimg, xx = hw % p.Wimg;

  __builtin_prefetch(s0.x, 0, 0);               // global_prefetch_b8

  f32x8 accv[MT];
#pragma unroll
  for (int mt = 0; mt < MT; ++mt)
#pragma unroll
    for (int e = 0; e < 8; ++e) accv[mt][e] = 0.f;

#pragma unroll
  for (int tap = 0; tap < TAPS; ++tap) {
    const int ty = (TAPS == 9) ? (tap / 3 - 1) : 0;
    const int tx = (TAPS == 9) ? (tap % 3 - 1) : 0;
    const int sy = yy + ty, sx = xx + tx;
    const bool inb = (sy >= 0) && (sy < Hi) && (sx >= 0) && (sx < p.Wimg);
    const int shw = sy * p.Wimg + sx;

    const float* bas0 = s0.x + (long)b * s0.sB + shw;
    const float* bas1 = (C1 > 0) ? (s1.x + (long)b * s1.sB + shw) : bas0;
    const float* sb0  = SC0 ? (s0.s + (long)b * s0.ssB + shw) : nullptr;
    const float* sb1  = SC1 ? (s1.s + (long)b * s1.ssB + shw) : nullptr;

#pragma unroll
    for (int kk = 0; kk < KS; ++kk) {
      // ---- B fragment: 32(K) x 16(N) bf16 gathered from concat(s0, s1)
      bf16x16 bfr;
#pragma unroll
      for (int e = 0; e < 16; ++e) {
        const int kl = kk * 32 + kmapK(e, 0);   // compile-time
        const int kh = kk * 32 + kmapK(e, 1);   // compile-time
        const int k  = hi ? kh : kl;            // one select
        const bool ok   = inb && (k < CIN);
        const bool use1 = (C1 > 0) && (k >= C0);
        const int  c    = use1 ? (k - C0) : k;
        const float* xp = use1 ? bas1 : bas0;
        const long sc   = use1 ? s1.sC : s0.sC;
        const long off  = ok ? (long)c * sc : 0;  // clamp keeps load in-bounds
        float v = xp[off];
        if (SC0 || SC1) {
          const bool dos = use1 ? SC1 : SC0;
          if (dos) {
            const float* sp = use1 ? sb1 : sb0;
            const long ssc  = use1 ? s1.ssC : s0.ssC;
            const long soff = ok ? (long)c * ssc : 0;
            v *= sp[soff];
          }
        }
        v = ok ? v : 0.f;
        bfr[e] = f2bf(v);
      }
      // ---- A fragments from LDS + WMMA
#pragma unroll
      for (int mt = 0; mt < MT; ++mt) {
        bf16x16 afr = wlds[((tap * KS + kk) * MT + mt) * 32 + lane];
        accv[mt] = __builtin_amdgcn_wmma_f32_16x16x32_bf16(
            false, afr, false, bfr, (short)0, accv[mt], false, false);
      }
    }
  }

  // ---- stats (LDS float atomics) + store
  // C/D layout: elem r -> M = mt*16 + r + 8*hi, N = ln
#pragma unroll
  for (int mt = 0; mt < MT; ++mt) {
#pragma unroll
    for (int r = 0; r < 8; ++r) {
      int ch = mt * 16 + r + hi * 8;
      bool vld = ch < p.Cout;
      float v = accv[mt][r];
      if (STATS) {
        if (vld) {
          atomicAdd(&bstat[ch],      v);       // ds_add_f32
          atomicAdd(&bstat[64 + ch], v * v);   // ds_add_f32
        }
      }
      if (vld) {
        if (p.bias) v += p.bias[ch];
        if (ACT == 1)      v = 1.f / (1.f + __expf(-v));
        else if (ACT == 2) v = tanhf(v);
        float* dst = p.Y + (long)ch * p.ysC + (long)b * p.ysB + hw;
        if (p.acc) v += *dst;
        *dst = v;
      }
    }
  }

  if (STATS) {
    __syncthreads();                            // all waves reach this
    if (tid < p.Cout) {
      atomicAdd(&p.stats[tid],      bstat[tid]);
      atomicAdd(&p.stats[64 + tid], bstat[64 + tid]);
    }
  }
}

// ---- per-pixel sigmoid attention over concat of up to 4 ten-channel nodes
struct AttP {
  const float *x0, *x1, *x2, *x3; int nsrc;
  const float* w; const float* b;
  float* out; int mode;              // 0: sigmoid(a), 1: 2 - sigmoid(a)
  int B, HW;
};
__global__ void k_att1(AttP p) {
  long idx = (long)blockIdx.x * blockDim.x + threadIdx.x;
  long P = (long)p.B * p.HW;
  if (idx >= P) return;
  int b = (int)(idx / p.HW), hw = (int)(idx % p.HW);
  const float* xs[4] = {p.x0, p.x1, p.x2, p.x3};
  float a = p.b[0];
  int k = 0;
  for (int s = 0; s < p.nsrc; ++s) {
    const float* x = xs[s];
    for (int c = 0; c < 10; ++c, ++k)
      a += p.w[k] * x[(long)b * 10 * p.HW + (long)c * p.HW + hw];
  }
  float sg = 1.f / (1.f + __expf(-a));
  p.out[idx] = p.mode ? (2.f - sg) : sg;
}

// ---- 1x1 conv 10->A + channel softmax ([B,A,HW] output)
struct DattP {
  const float* x; const float* w; const float* b; int A;
  float* out; int B, HW;
};
__global__ void k_datt(DattP p) {
  long idx = (long)blockIdx.x * blockDim.x + threadIdx.x;
  long P = (long)p.B * p.HW;
  if (idx >= P) return;
  int b = (int)(idx / p.HW), hw = (int)(idx % p.HW);
  float v[5]; float mx = -1e30f;
  for (int a = 0; a < p.A; ++a) {
    float s = p.b[a];
    for (int c = 0; c < 10; ++c)
      s += p.w[a * 10 + c] * p.x[(long)b * 10 * p.HW + (long)c * p.HW + hw];
    v[a] = s; mx = fmaxf(mx, s);
  }
  float sum = 0.f;
  for (int a = 0; a < p.A; ++a) { v[a] = __expf(v[a] - mx); sum += v[a]; }
  float inv = 1.f / sum;
  for (int a = 0; a < p.A; ++a)
    p.out[((long)b * p.A + a) * p.HW + hw] = v[a] * inv;
}

__global__ void k_bnfin(const float* stats, const float* g, const float* bb,
                        float* ss, int C, float invN) {
  int c = threadIdx.x;
  if (c >= C) return;
  float mu  = stats[c] * invN;
  float var = stats[64 + c] * invN - mu * mu;
  float sc  = g[c] * rsqrtf(var + 1e-5f);
  ss[c] = sc;
  ss[64 + c] = bb[c] - mu * sc;
}

__global__ void k_bnrelu(const float* x, const float* ss, float* y,
                         long ysC, long ysB, int C, int B, int HW, int acc) {
  long n = (long)C * B * HW;
  long idx = (long)blockIdx.x * blockDim.x + threadIdx.x;
  if (idx >= n) return;
  long P = (long)B * HW;
  int  c = (int)(idx / P);
  long rem = idx % P;
  int  b = (int)(rem / HW), hw = (int)(rem % HW);
  float v = fmaxf(0.f, x[idx] * ss[c] + ss[64 + c]);
  float* d = y + (long)c * ysC + (long)b * ysB + hw;
  if (acc) v += *d;
  *d = v;
}

// out = (1-z)*h + z*cnm   (z = gates channels 10..19, [20,P] layout)
__global__ void k_grufin(const float* gates, const float* cn, const float* h,
                         long hsB, float* out, int B, int HW) {
  long P = (long)B * HW;
  long n = 10 * P;
  long idx = (long)blockIdx.x * blockDim.x + threadIdx.x;
  if (idx >= n) return;
  int  c = (int)(idx / P);
  long rem = idx % P;
  int  b = (int)(rem / HW), hw = (int)(rem % HW);
  float z  = gates[(long)(10 + c) * P + rem];
  float cm = cn[idx];
  float hv = h[(long)b * hsB + (long)c * HW + hw];
  out[(long)b * 10 * HW + (long)c * HW + hw] = (1.f - z) * hv + z * cm;
}

// ---------------------------------------------------------------------------
// Host orchestration
// ---------------------------------------------------------------------------

struct Pool {
  float *T1, *T2, *T3, *CN, *MSG, *ST, *SS;
  int B, HW, Wimg;
};
struct RelW { const float *w1, *g1, *b1, *w2, *g2, *b2; };

static inline dim3 gridN(long n) { return dim3((unsigned)((n + 255) / 256)); }

static void launch_gemm(hipStream_t st, const GemmP& p) {
  int tiles = (p.B * p.HW) / 16;
  dim3 grid((unsigned)(tiles / 4));
  const bool sc0 = p.s0.s != nullptr;
  const bool sc1 = p.s1.s != nullptr;
  if (p.taps == 1) {
    if (p.C0 == 10 && p.C1 == 10) {            // relation conv1 (cat -> 20)
      if (sc0)      k_gemm<2, 10, 10, 1, true,  false, true, 0><<<grid, 128, 0, st>>>(p);
      else if (sc1) k_gemm<2, 10, 10, 1, false, true,  true, 0><<<grid, 128, 0, st>>>(p);
      else          k_gemm<2, 10, 10, 1, false, false, true, 0><<<grid, 128, 0, st>>>(p);
    } else if (p.C0 == 20) {                   // relation conv2 (20 -> 10)
      k_gemm<1, 20, 0, 1, false, false, true, 0><<<grid, 128, 0, st>>>(p);
    } else {                                   // ctx projection (256 -> 10)
      k_gemm<1, 256, 0, 1, true, false, true, 0><<<grid, 128, 0, st>>>(p);
    }
  } else {
    if (p.Cout == 20)                          // GRU gates 3x3 (cat -> 20), sigmoid
      k_gemm<2, 10, 10, 9, false, false, false, 1><<<grid, 128, 0, st>>>(p);
    else                                       // GRU candidate 3x3 (cat[msg,r*h]), tanh
      k_gemm<1, 10, 10, 9, false, true, false, 2><<<grid, 128, 0, st>>>(p);
  }
}

static void run_relation(hipStream_t st, const RelW& rw, const Src& s0,
                         const Src& s1, float* dst, int accflag, const Pool& q) {
  long P = (long)q.B * q.HW;
  float invN = 1.f / (float)P;
  // conv1 (cat -> 20) + stats
  hipMemsetAsync(q.ST, 0, 128 * sizeof(float), st);
  GemmP g1{};
  g1.W = rw.w1; g1.bias = nullptr; g1.Cout = 20; g1.C0 = 10; g1.C1 = 10; g1.taps = 1;
  g1.s0 = s0; g1.s1 = s1;
  g1.Y = q.T1; g1.ysC = P; g1.ysB = q.HW; g1.stats = q.ST; g1.acc = 0; g1.act = 0;
  g1.B = q.B; g1.HW = q.HW; g1.Wimg = q.Wimg;
  launch_gemm(st, g1);
  k_bnfin<<<1, 32, 0, st>>>(q.ST, rw.g1, rw.b1, q.SS, 20, invN);
  k_bnrelu<<<gridN(20 * P), 256, 0, st>>>(q.T1, q.SS, q.T2, P, (long)q.HW, 20, q.B, q.HW, 0);
  // conv2 (20 -> 10) + stats
  hipMemsetAsync(q.ST, 0, 128 * sizeof(float), st);
  GemmP g2{};
  g2.W = rw.w2; g2.bias = nullptr; g2.Cout = 10; g2.C0 = 20; g2.C1 = 0; g2.taps = 1;
  g2.s0 = Src{q.T2, P, (long)q.HW, nullptr, 0, 0};
  g2.s1 = Src{nullptr, 0, 0, nullptr, 0, 0};
  g2.Y = q.T3; g2.ysC = P; g2.ysB = q.HW; g2.stats = q.ST; g2.acc = 0; g2.act = 0;
  g2.B = q.B; g2.HW = q.HW; g2.Wimg = q.Wimg;
  launch_gemm(st, g2);
  k_bnfin<<<1, 32, 0, st>>>(q.ST, rw.g2, rw.b2, q.SS, 10, invN);
  k_bnrelu<<<gridN(10 * P), 256, 0, st>>>(q.T3, q.SS, dst, P, (long)q.HW, 10, q.B, q.HW, accflag);
}

static void run_gru(hipStream_t st, const float* wg, const float* bg,
                    const float* wc, const float* bc, const float* h,
                    float* out, const Pool& q) {
  long P = (long)q.B * q.HW;
  // gates = sigmoid(conv3x3(cat[msg, h]))
  GemmP gg{};
  gg.W = wg; gg.bias = bg; gg.Cout = 20; gg.C0 = 10; gg.C1 = 10; gg.taps = 9;
  gg.s0 = Src{q.MSG, P, (long)q.HW, nullptr, 0, 0};
  gg.s1 = Src{h, (long)q.HW, (long)10 * q.HW, nullptr, 0, 0};
  gg.Y = q.T1; gg.ysC = P; gg.ysB = q.HW; gg.stats = nullptr; gg.acc = 0; gg.act = 1;
  gg.B = q.B; gg.HW = q.HW; gg.Wimg = q.Wimg;
  launch_gemm(st, gg);
  // cnm = tanh(conv3x3(cat[msg, r*h])), r = gates ch 0..9
  GemmP gc{};
  gc.W = wc; gc.bias = bc; gc.Cout = 10; gc.C0 = 10; gc.C1 = 10; gc.taps = 9;
  gc.s0 = Src{q.MSG, P, (long)q.HW, nullptr, 0, 0};
  gc.s1 = Src{h, (long)q.HW, (long)10 * q.HW, q.T1, P, (long)q.HW};
  gc.Y = q.CN; gc.ysC = P; gc.ysB = q.HW; gc.stats = nullptr; gc.acc = 0; gc.act = 2;
  gc.B = q.B; gc.HW = q.HW; gc.Wimg = q.Wimg;
  launch_gemm(st, gc);
  k_grufin<<<gridN(10 * P), 256, 0, st>>>(q.T1, q.CN, h, (long)10 * q.HW, out, q.B, q.HW);
}

extern "C" void kernel_launch(void* const* d_in, const int* in_sizes, int n_in,
                              void* d_out, int out_size, void* d_ws, size_t ws_size,
                              hipStream_t stream) {
  const int B = 8, Wimg = 96, HW = 96 * 96;
  const long P = (long)B * HW;
  const long nodeSz = (long)B * 10 * HW;
  float* ws = (float*)d_ws;
  if (ws_size < (size_t)(85 * P + 256) * sizeof(float)) return;

  const float* f_node    = (const float*)d_in[0];
  const float* h_nodes   = (const float*)d_in[1];
  const float* p_nodes   = (const float*)d_in[2];
  const float* p_fea     = (const float*)d_in[3];
  const float* cattf_w   = (const float*)d_in[4];
  const float* cattf_b   = (const float*)d_in[5];
  const float* cattu_w   = (const float*)d_in[6];
  const float* cattu_b   = (const float*)d_in[7];
  const float* cattl_w   = (const float*)d_in[8];
  const float* cattl_b   = (const float*)d_in[9];
  const float* datt_h_w  = (const float*)d_in[10];
  const float* datt_h_b  = (const float*)d_in[11];
  const float* datt_u_w  = (const float*)d_in[12];
  const float* datt_u_b  = (const float*)d_in[13];
  const float* datt_l_w  = (const float*)d_in[14];
  const float* datt_l_b  = (const float*)d_in[15];
  const float* rel_w1    = (const float*)d_in[16];
  const float* rel_g1    = (const float*)d_in[17];
  const float* rel_b1    = (const float*)d_in[18];
  const float* rel_w2    = (const float*)d_in[19];
  const float* rel_g2    = (const float*)d_in[20];
  const float* rel_b2    = (const float*)d_in[21];
  const float* ctx_att_w = (const float*)d_in[22];
  const float* ctx_att_b = (const float*)d_in[23];
  const float* ctx_proj_w= (const float*)d_in[24];
  const float* ctx_g     = (const float*)d_in[25];
  const float* ctx_b     = (const float*)d_in[26];
  const float* gru_wg    = (const float*)d_in[27];
  const float* gru_bg    = (const float*)d_in[28];
  const float* gru_wc    = (const float*)d_in[29];
  const float* gru_bc    = (const float*)d_in[30];
  (void)in_sizes; (void)n_in; (void)out_size;

  const float* h0 = h_nodes;
  const float* h1 = h_nodes + nodeSz;
  auto part = [&](int i) { return p_nodes + (long)i * nodeSz; };
  float* out = (float*)d_out;

  // workspace layout (floats)
  float* A0 = ws;            // comp_map_f        [P]
  float* A1 = ws + P;        // comp_map_u        [P]
  float* A2 = ws + 2 * P;    // comp_map_l        [P]
  float* A3 = ws + 3 * P;    // decomp_att (3)    [3P]
  float* A4 = ws + 6 * P;    // att_u (5)         [5P]
  float* A5 = ws + 11 * P;   // att_l (3)         [3P]
  float* A6 = ws + 14 * P;   // ctx map (2-a)     [P]
  Pool q;
  q.T1 = ws + 15 * P; q.T2 = ws + 35 * P; q.T3 = ws + 55 * P;
  q.CN = ws + 65 * P; q.MSG = ws + 75 * P;
  q.ST = ws + 85 * P; q.SS = q.ST + 128;
  q.B = B; q.HW = HW; q.Wimg = Wimg;

  const float invN = 1.f / (float)P;
  const long sC = HW, sB = 10 * (long)HW;   // node tensor strides

  // ===================== full graph =====================
  {
    AttP a{}; a.x0 = h0; a.x1 = h1; a.nsrc = 2; a.w = cattf_w; a.b = cattf_b;
    a.out = A0; a.mode = 0; a.B = B; a.HW = HW;
    k_att1<<<gridN(P), 256, 0, stream>>>(a);
  }
  RelW R0{rel_w1, rel_g1, rel_b1, rel_w2, rel_g2, rel_b2};
  RelW R1{rel_w1 + 400,  rel_g1 + 20, rel_b1 + 20, rel_w2 + 200, rel_g2 + 10, rel_b2 + 10};
  RelW R2{rel_w1 + 800,  rel_g1 + 40, rel_b1 + 40, rel_w2 + 400, rel_g2 + 20, rel_b2 + 20};
  RelW R3{rel_w1 + 1200, rel_g1 + 60, rel_b1 + 60, rel_w2 + 600, rel_g2 + 30, rel_b2 + 30};

  Src sF{f_node, sC, sB, nullptr, 0, 0};
  run_relation(stream, R0, sF, Src{h0, sC, sB, A0, 0, (long)HW}, q.MSG, 0, q);
  run_relation(stream, R0, sF, Src{h1, sC, sB, A0, 0, (long)HW}, q.MSG, 1, q);
  run_gru(stream, gru_wg, gru_bg, gru_wc, gru_bc, f_node, out, q);

  // ===================== half graphs =====================
  {
    DattP d{}; d.x = f_node; d.w = datt_h_w; d.b = datt_h_b; d.A = 3;
    d.out = A3; d.B = B; d.HW = HW;
    k_datt<<<gridN(P), 256, 0, stream>>>(d);
  }
  // upper half
  run_relation(stream, R1, Src{f_node, sC, sB, A3 + 1 * HW, 0, 3L * HW},
               Src{h0, sC, sB, nullptr, 0, 0}, q.MSG, 0, q);
  {
    AttP a{}; a.x0 = part(0); a.x1 = part(1); a.x2 = part(2); a.x3 = part(3);
    a.nsrc = 4; a.w = cattu_w; a.b = cattu_b; a.out = A1; a.mode = 0; a.B = B; a.HW = HW;
    k_att1<<<gridN(P), 256, 0, stream>>>(a);
  }
  for (int j = 0; j < 4; ++j)
    run_relation(stream, R2, Src{h0, sC, sB, nullptr, 0, 0},
                 Src{part(j), sC, sB, A1, 0, (long)HW}, q.MSG, 1, q);
  run_gru(stream, gru_wg + 1 * 3600, gru_bg + 1 * 20, gru_wc + 1 * 1800,
          gru_bc + 1 * 10, h0, out + 1 * nodeSz, q);
  // lower half
  run_relation(stream, R1, Src{f_node, sC, sB, A3 + 2 * HW, 0, 3L * HW},
               Src{h1, sC, sB, nullptr, 0, 0}, q.MSG, 0, q);
  {
    AttP a{}; a.x0 = part(4); a.x1 = part(5); a.nsrc = 2;
    a.w = cattl_w; a.b = cattl_b; a.out = A2; a.mode = 0; a.B = B; a.HW = HW;
    k_att1<<<gridN(P), 256, 0, stream>>>(a);
  }
  for (int j = 4; j < 6; ++j)
    run_relation(stream, R2, Src{h1, sC, sB, nullptr, 0, 0},
                 Src{part(j), sC, sB, A2, 0, (long)HW}, q.MSG, 1, q);
  run_gru(stream, gru_wg + 2 * 3600, gru_bg + 2 * 20, gru_wc + 2 * 1800,
          gru_bc + 2 * 10, h1, out + 2 * nodeSz, q);

  // ===================== part graph =====================
  {
    DattP d{}; d.x = h0; d.w = datt_u_w; d.b = datt_u_b; d.A = 5;
    d.out = A4; d.B = B; d.HW = HW;
    k_datt<<<gridN(P), 256, 0, stream>>>(d);
    DattP e{}; e.x = h1; e.w = datt_l_w; e.b = datt_l_b; e.A = 3;
    e.out = A5; e.B = B; e.HW = HW;
    k_datt<<<gridN(P), 256, 0, stream>>>(e);
  }
  const int nbr[6][2] = {{1,5},{0,2},{1,3},{2,4},{3,5},{0,4}};  // PART_LIST
  for (int i = 0; i < 6; ++i) {
    // decomposition message
    if (i < 4)
      run_relation(stream, R3, Src{h0, sC, sB, A4 + (long)(i + 1) * HW, 0, 5L * HW},
                   Src{part(i), sC, sB, nullptr, 0, 0}, q.MSG, 0, q);
    else
      run_relation(stream, R3, Src{h1, sC, sB, A5 + (long)(i - 3) * HW, 0, 3L * HW},
                   Src{part(i), sC, sB, nullptr, 0, 0}, q.MSG, 0, q);
    // contextual dependency: a = sigmoid(ctx_att(cat[neighbors])), map = 2-a
    {
      AttP a{}; a.x0 = part(nbr[i][0]); a.x1 = part(nbr[i][1]); a.nsrc = 2;
      a.w = ctx_att_w + i * 20; a.b = ctx_att_b + i;
      a.out = A6; a.mode = 1; a.B = B; a.HW = HW;
      k_att1<<<gridN(P), 256, 0, stream>>>(a);
    }
    hipMemsetAsync(q.ST, 0, 128 * sizeof(float), stream);
    GemmP gd{};
    gd.W = ctx_proj_w + (long)i * 2560; gd.bias = nullptr;
    gd.Cout = 10; gd.C0 = 256; gd.C1 = 0; gd.taps = 1;
    gd.s0 = Src{p_fea, (long)HW, 256L * HW, A6, 0, (long)HW};
    gd.s1 = Src{nullptr, 0, 0, nullptr, 0, 0};
    gd.Y = q.T3; gd.ysC = P; gd.ysB = HW; gd.stats = q.ST; gd.acc = 0; gd.act = 0;
    gd.B = B; gd.HW = HW; gd.Wimg = Wimg;
    launch_gemm(stream, gd);
    k_bnfin<<<1, 32, 0, stream>>>(q.ST, ctx_g + i * 10, ctx_b + i * 10, q.SS, 10, invN);
    k_bnrelu<<<gridN(10 * P), 256, 0, stream>>>(q.T3, q.SS, q.MSG, P, (long)HW, 10, B, HW, 1);
    // GRU update
    run_gru(stream, gru_wg + (long)(3 + i) * 3600, gru_bg + (long)(3 + i) * 20,
            gru_wc + (long)(3 + i) * 1800, gru_bc + (long)(3 + i) * 10,
            part(i), out + (long)(3 + i) * nodeSz, q);
  }
}
